// GatedMultiheadAttention_5987184410925
// MI455X (gfx1250) — matrix operand order repaired and checked by reading, not compiled, not run on hardware
//
#include <hip/hip_runtime.h>
#include <hip/hip_bf16.h>

// ---------------------------------------------------------------------------
// Gated multi-head attention for gfx1250 (MI455X), wave32 + WMMA f16->f32.
//   1) cvt_f32_f16 : inputs/weights -> f16 workspace
//   2) gemm_xwT    : Q/K -> [B,H,N,D] f16, V -> [B,H,D,N] f16 (transposed)
//                    64x64 WMMA tile per wave (1 b128 load per WMMA)
//   3) attn_fwd    : flash attention; K/V tiles double-buffered in LDS via
//                    async global->LDS (ASYNCcnt) shared by all 8 waves
//   4) gemm_xwT    : out @ WoT + bo -> f32 d_out
// ---------------------------------------------------------------------------

typedef __attribute__((ext_vector_type(16))) _Float16 v16h;
typedef __attribute__((ext_vector_type(8)))  float    v8f;
typedef __attribute__((ext_vector_type(4)))  _Float16 v4h;
typedef __attribute__((ext_vector_type(4)))  int      v4i;

#define EMBED  1024
#define NH     16
#define HD     64
#define BATCH  2
#define SEQ    2048
#define TOKENS (BATCH * SEQ)   // 4096

#ifndef __has_builtin
#define __has_builtin(x) 0
#endif
#if __has_builtin(__builtin_amdgcn_global_load_async_to_lds_b128) && \
    __has_builtin(__builtin_amdgcn_s_wait_asynccnt)
#define HAVE_ASYNC_LDS 1
#else
#define HAVE_ASYNC_LDS 0
#endif

// ---- WMMA fragment loaders (contiguous k = 16*half + s for both A and B, wave32) ---------------
// A (16x32 f16): M = lane&15; lanes 0-15 hold K 0..15,
// lanes 16-31 hold K 16..31. Two contiguous 16B loads.
__device__ __forceinline__ v16h load_a_frag(const _Float16* __restrict__ A,
                                            int lda, int row0, int k0) {
  const int lane = threadIdx.x & 31;
  const int m = lane & 15, hf = lane >> 4;
  const _Float16* p = A + (size_t)(row0 + m) * lda + k0 + 16 * hf;
  v16h r;
  ((uint4*)&r)[0] = *(const uint4*)(p);
  ((uint4*)&r)[1] = *(const uint4*)(p + 8);
  return r;
}

// B (32x16 f16): N = lane&15; lanes 0-15 hold K 0..15, lanes 16-31 K 16..31.
// Source is [N,K] row-major -> one 32B contiguous region per lane.
__device__ __forceinline__ v16h load_b_frag(const _Float16* __restrict__ Wm,
                                            int ldw, int col0, int k0) {
  const int lane = threadIdx.x & 31;
  const int n = lane & 15, hf = lane >> 4;
  const _Float16* p = Wm + (size_t)(col0 + n) * ldw + k0 + 16 * hf;
  v16h r;
  ((uint4*)&r)[0] = *(const uint4*)(p);
  ((uint4*)&r)[1] = *(const uint4*)(p + 8);
  return r;
}

// Same B-fragment gather but from an LDS tile (emits ds_load_b128).
__device__ __forceinline__ v16h lds_b_frag(const _Float16* base, int pitch,
                                           int col0, int k0) {
  const int lane = threadIdx.x & 31;
  const int n = lane & 15, hf = lane >> 4;
  const _Float16* p = base + (col0 + n) * pitch + k0 + 16 * hf;
  v16h r;
  ((uint4*)&r)[0] = *(const uint4*)(p);
  ((uint4*)&r)[1] = *(const uint4*)(p + 8);
  return r;
}

__device__ __forceinline__ v8f wmma_f16(v16h a, v16h b, v8f c) {
  return __builtin_amdgcn_wmma_f32_16x16x32_f16(false, a, false, b,
                                                (short)0, c, false, false);
}

// 16B global -> LDS copy: async (ASYNCcnt-tracked) when available.
// Builtin signature (per hipcc diagnostic): (v4i AS1*, v4i AS3*, imm, imm).
__device__ __forceinline__ void cp_g2l_16B(const _Float16* gsrc, _Float16* ldst) {
#if HAVE_ASYNC_LDS
  __builtin_amdgcn_global_load_async_to_lds_b128(
      (__attribute__((address_space(1))) v4i*)gsrc,
      (__attribute__((address_space(3))) v4i*)ldst, 0, 0);
#else
  *(uint4*)ldst = *(const uint4*)gsrc;
#endif
}

__device__ __forceinline__ void wait_async_fills() {
#if HAVE_ASYNC_LDS
  __builtin_amdgcn_s_wait_asynccnt(0);
#endif
}

// ---- f32 -> f16 conversion ------------------------------------------------
__global__ __launch_bounds__(256) void cvt_f32_f16(const float* __restrict__ in,
                                                   _Float16* __restrict__ out,
                                                   int n) {
  int i = (blockIdx.x * 256 + threadIdx.x) * 4;
  if (i < n) {
    float4 v = *(const float4*)(in + i);
    v4h h;
    h.x = (_Float16)v.x; h.y = (_Float16)v.y;
    h.z = (_Float16)v.z; h.w = (_Float16)v.w;
    *(v4h*)(out + i) = h;
  }
}

// ---- GEMM: C[m,n] = sum_k A[m,k]*W[n,k] + bias[n] -------------------------
// 128 threads = 4 waves; each wave owns a 64x64 tile (4x4 WMMA accumulators,
// 1 b128 load per WMMA). A/B slabs are L2-resident (192MB L2 >> working set).
// mode 0: f16 head-split [B,H,SEQ,HD]; mode 1: f16 [B,H,HD,SEQ]; mode 2: f32.
__global__ __launch_bounds__(128) void gemm_xwT(
    const _Float16* __restrict__ A, const _Float16* __restrict__ Wm,
    const float* __restrict__ bias, void* __restrict__ outp,
    int M, int N, int K, int mode) {
  const int tilesN = N >> 7;
  const int bx = blockIdx.x % tilesN;
  const int by = blockIdx.x / tilesN;
  const int w  = threadIdx.x >> 5;
  const int row0 = by * 128 + (w >> 1) * 64;
  const int col0 = bx * 128 + (w & 1) * 64;

  v8f acc[4][4] = {};
  for (int k0 = 0; k0 < K; k0 += 32) {
    if (k0 + 32 < K) {  // global_prefetch_b8 hint for the next K slab
      __builtin_prefetch(A  + (size_t)row0 * K + k0 + 32, 0, 1);
      __builtin_prefetch(Wm + (size_t)col0 * K + k0 + 32, 0, 1);
    }
    v16h a[4], b[4];
#pragma unroll
    for (int i = 0; i < 4; ++i) a[i] = load_a_frag(A, K, row0 + 16 * i, k0);
#pragma unroll
    for (int i = 0; i < 4; ++i) b[i] = load_b_frag(Wm, K, col0 + 16 * i, k0);
#pragma unroll
    for (int i = 0; i < 4; ++i)
#pragma unroll
      for (int j = 0; j < 4; ++j) acc[i][j] = wmma_f16(a[i], b[j], acc[i][j]);
  }

  const int lane = threadIdx.x & 31;
  const int nn = lane & 15, hh = lane >> 4;
#pragma unroll
  for (int i = 0; i < 4; ++i)
#pragma unroll
    for (int j = 0; j < 4; ++j)
#pragma unroll
      for (int r = 0; r < 8; ++r) {
        int m = row0 + 16 * i + r + 8 * hh;   // C layout: M = vgpr + 8*half
        int n = col0 + 16 * j + nn;           //           N = lane&15
        float v = acc[i][j][r] + bias[n];
        if (mode == 2) {
          ((float*)outp)[(size_t)m * N + n] = v;
        } else {
          int b2 = m >> 11, mm = m & (SEQ - 1);
          int hD = n >> 6, d = n & (HD - 1);
          size_t idx = (mode == 0)
              ? ((size_t)((b2 * NH + hD) * SEQ + mm) * HD + d)
              : ((size_t)((b2 * NH + hD) * HD + d) * SEQ + mm);
          ((_Float16*)outp)[idx] = (_Float16)v;
        }
      }
}

// ---- Flash attention: 8 waves/block share one (b,h); K/V tiles live in ----
// ---- LDS, double-buffered via async global->LDS fills ---------------------
#define KP 72   // K-tile LDS pitch (64 + 8 pad halfs)
#define VP 40   // V-tile LDS pitch (32 + 8 pad halfs)

__global__ __launch_bounds__(256) void attn_fwd(
    const _Float16* __restrict__ Q,   // [BH, SEQ, HD]
    const _Float16* __restrict__ Kh,  // [BH, SEQ, HD]
    const _Float16* __restrict__ Vt,  // [BH, HD, SEQ]
    const float* __restrict__ gate,
    _Float16* __restrict__ Omid)      // [B*SEQ, EMBED]
{
  __shared__ __align__(16) _Float16 Klds[2][32][KP];  // 32 keys x 64 d
  __shared__ __align__(16) _Float16 Vlds[2][64][VP];  // 64 d x 32 keys
  __shared__ __align__(16) _Float16 pshm[8][16][40];  // per-wave P staging

  const int t     = threadIdx.x;
  const int wslot = t >> 5;
  const int bh    = blockIdx.x >> 4;           // 16 blocks per (b,h)
  const int qtile = (blockIdx.x & 15) * 8 + wslot;
  const int b = bh >> 4, h = bh & 15;
  const int lane = t & 31;
  const int nn = lane & 15, hh = lane >> 4;

  const _Float16* Qb = Q  + (size_t)bh * SEQ * HD + (size_t)qtile * 16 * HD;
  const _Float16* Kb = Kh + (size_t)bh * SEQ * HD;
  const _Float16* Vb = Vt + (size_t)bh * HD * SEQ;

  // per-head sigmoid gate folded into logit scale (1/sqrt(64) = 0.125)
  const float sc = (1.f / (1.f + __expf(-gate[h]))) * 0.125f;

  const v16h qf0 = load_a_frag(Qb, HD, 0, 0);
  const v16h qf1 = load_a_frag(Qb, HD, 0, 32);

  // cooperative tile staging: 256 threads x 16B each per tile
  const int kr = t >> 3, kc = (t & 7) * 8;   // K tile: 32 rows x 64 halfs
  const int vr = t >> 2, vc = (t & 3) * 8;   // V tile: 64 rows x 32 halfs

  // prologue: fill buffer 0 with tile j=0
  cp_g2l_16B(Kb + (size_t)kr * HD + kc, &Klds[0][kr][kc]);
  cp_g2l_16B(Vb + (size_t)vr * SEQ + vc, &Vlds[0][vr][vc]);

  float mrow[8], lrow[8];
  v8f o[4] = {};
#pragma unroll
  for (int r = 0; r < 8; ++r) { mrow[r] = -1e30f; lrow[r] = 0.f; }

  int buf = 0;
  for (int j = 0; j < SEQ; j += 32, buf ^= 1) {
    wait_async_fills();      // my fills done
    __syncthreads();         // everyone's fills visible

    if (j + 32 < SEQ) {      // async-fill the other buffer for next tile
      cp_g2l_16B(Kb + (size_t)(j + 32 + kr) * HD + kc, &Klds[buf ^ 1][kr][kc]);
      cp_g2l_16B(Vb + (size_t)vr * SEQ + (j + 32) + vc, &Vlds[buf ^ 1][vr][vc]);
    }

    // logits S[16q x 32k]: two 16x16 C tiles, d reduced in two WMMA steps
    const _Float16* Kt = &Klds[buf][0][0];
    v8f s0 = {}, s1 = {};
    s0 = wmma_f16(qf0, lds_b_frag(Kt, KP, 0,  0),  s0);
    s0 = wmma_f16(qf1, lds_b_frag(Kt, KP, 0,  32), s0);
    s1 = wmma_f16(qf0, lds_b_frag(Kt, KP, 16, 0),  s1);
    s1 = wmma_f16(qf1, lds_b_frag(Kt, KP, 16, 32), s1);

    float p0[8], p1[8], alpha[8];
#pragma unroll
    for (int r = 0; r < 8; ++r) {
      float x0 = s0[r] * sc, x1 = s1[r] * sc;
      float vmx = fmaxf(x0, x1);               // row-max over 16-lane half
      vmx = fmaxf(vmx, __shfl_xor(vmx, 1, 32));
      vmx = fmaxf(vmx, __shfl_xor(vmx, 2, 32));
      vmx = fmaxf(vmx, __shfl_xor(vmx, 4, 32));
      vmx = fmaxf(vmx, __shfl_xor(vmx, 8, 32));
      float mnew = fmaxf(mrow[r], vmx);
      alpha[r] = __expf(mrow[r] - mnew);
      p0[r] = __expf(x0 - mnew);
      p1[r] = __expf(x1 - mnew);
      float rs = p0[r] + p1[r];
      rs += __shfl_xor(rs, 1, 32);
      rs += __shfl_xor(rs, 2, 32);
      rs += __shfl_xor(rs, 4, 32);
      rs += __shfl_xor(rs, 8, 32);
      lrow[r] = lrow[r] * alpha[r] + rs;
      mrow[r] = mnew;
    }
#pragma unroll
    for (int c = 0; c < 4; ++c)
#pragma unroll
      for (int r = 0; r < 8; ++r) o[c][r] *= alpha[r];

    // C-layout P -> LDS -> A-fragment layout (per-wave region; DS in-order)
#pragma unroll
    for (int r = 0; r < 8; ++r) {
      pshm[wslot][r + 8 * hh][nn]      = (_Float16)p0[r];
      pshm[wslot][r + 8 * hh][16 + nn] = (_Float16)p1[r];
    }
    v16h pf;
    ((uint4*)&pf)[0] = *(const uint4*)&pshm[wslot][nn][16 * hh];
    ((uint4*)&pf)[1] = *(const uint4*)&pshm[wslot][nn][16 * hh + 8];

    const _Float16* Vtb = &Vlds[buf][0][0];
#pragma unroll
    for (int c = 0; c < 4; ++c)
      o[c] = wmma_f16(pf, lds_b_frag(Vtb, VP, c * 16, 0), o[c]);

    __syncthreads();         // all reads done before buffer is refilled
  }

  float rinv[8];
#pragma unroll
  for (int r = 0; r < 8; ++r) rinv[r] = 1.f / lrow[r];
  const int qrow0 = qtile * 16;
#pragma unroll
  for (int c = 0; c < 4; ++c)
#pragma unroll
    for (int r = 0; r < 8; ++r) {
      int qr  = qrow0 + r + 8 * hh;
      int col = h * HD + c * 16 + nn;
      Omid[(size_t)(b * SEQ + qr) * EMBED + col] = (_Float16)(o[c][r] * rinv[r]);
    }
}

// ---------------------------------------------------------------------------
extern "C" void kernel_launch(void* const* d_in, const int* in_sizes, int n_in,
                              void* d_out, int out_size, void* d_ws, size_t ws_size,
                              hipStream_t stream) {
  (void)in_sizes; (void)n_in; (void)out_size; (void)ws_size;
  const float* q    = (const float*)d_in[0];
  const float* k    = (const float*)d_in[1];
  const float* v    = (const float*)d_in[2];
  const float* Wq   = (const float*)d_in[3];
  const float* bq   = (const float*)d_in[4];
  const float* Wk   = (const float*)d_in[5];
  const float* bk   = (const float*)d_in[6];
  const float* Wv   = (const float*)d_in[7];
  const float* bv   = (const float*)d_in[8];
  const float* Wo   = (const float*)d_in[9];
  const float* bo   = (const float*)d_in[10];
  const float* gate = (const float*)d_in[11];

  _Float16* ws = (_Float16*)d_ws;
  const size_t ACT = (size_t)TOKENS * EMBED;   // 4M halfs
  const size_t WSZ = (size_t)EMBED * EMBED;    // 1M halfs
  _Float16* Xq  = ws;
  _Float16* Xk  = Xq + ACT;
  _Float16* Xv  = Xk + ACT;
  _Float16* Qh  = Xv + ACT;
  _Float16* Kh  = Qh + ACT;
  _Float16* Vt  = Kh + ACT;
  _Float16* Om  = Vt + ACT;
  _Float16* Wq16 = Om + ACT;
  _Float16* Wk16 = Wq16 + WSZ;
  _Float16* Wv16 = Wk16 + WSZ;
  _Float16* Wo16 = Wv16 + WSZ;

  const int cgA = (int)(ACT / 4 / 256);
  const int cgW = (int)(WSZ / 4 / 256);
  cvt_f32_f16<<<cgA, 256, 0, stream>>>(q,  Xq,  (int)ACT);
  cvt_f32_f16<<<cgA, 256, 0, stream>>>(k,  Xk,  (int)ACT);
  cvt_f32_f16<<<cgA, 256, 0, stream>>>(v,  Xv,  (int)ACT);
  cvt_f32_f16<<<cgW, 256, 0, stream>>>(Wq, Wq16, (int)WSZ);
  cvt_f32_f16<<<cgW, 256, 0, stream>>>(Wk, Wk16, (int)WSZ);
  cvt_f32_f16<<<cgW, 256, 0, stream>>>(Wv, Wv16, (int)WSZ);
  cvt_f32_f16<<<cgW, 256, 0, stream>>>(Wo, Wo16, (int)WSZ);

  const dim3 gg((TOKENS / 128) * (EMBED / 128));  // 256 blocks of 4 waves
  gemm_xwT<<<gg, 128, 0, stream>>>(Xq, Wq16, bq, Qh, TOKENS, EMBED, EMBED, 0);
  gemm_xwT<<<gg, 128, 0, stream>>>(Xk, Wk16, bk, Kh, TOKENS, EMBED, EMBED, 0);
  gemm_xwT<<<gg, 128, 0, stream>>>(Xv, Wv16, bv, Vt, TOKENS, EMBED, EMBED, 1);

  attn_fwd<<<512, 256, 0, stream>>>(Qh, Kh, Vt, gate, Om);

  gemm_xwT<<<gg, 128, 0, stream>>>(Om, Wo16, bo, d_out, TOKENS, EMBED, EMBED, 2);
}
